// ConvGNN_3006477107598
// MI455X (gfx1250) — matrix operand order repaired and checked by reading, not compiled, not run on hardware
//
#include <hip/hip_runtime.h>

#define N0   131072
#define NE   1048576
#define INF0 45
#define HDIM 128
#define NLAY 4

typedef __attribute__((ext_vector_type(16))) _Float16 v16h;
typedef __attribute__((ext_vector_type(8)))  _Float16 v8h;
typedef __attribute__((ext_vector_type(4)))  _Float16 v4h;
typedef __attribute__((ext_vector_type(8)))  float    v8f;

static __device__ __forceinline__ void atomAddF(float* p, float v) {
    __hip_atomic_fetch_add(p, v, __ATOMIC_RELAXED, __HIP_MEMORY_SCOPE_AGENT);
}

// ---------------- fills ----------------
__global__ void fill_f32(float* p, float v, int n) {
    int i = blockIdx.x * blockDim.x + threadIdx.x;
    for (; i < n; i += gridDim.x * blockDim.x) p[i] = v;
}
__global__ void fill_i32(int* p, int v, int n) {
    int i = blockIdx.x * blockDim.x + threadIdx.x;
    for (; i < n; i += gridDim.x * blockDim.x) p[i] = v;
}

// ---------------- operand prep (f16, padded, W transposed) ----------------
// x0 [N0 x 45] f32 -> xh [N0 x 64] f16 zero-padded
__global__ void cvt_x0(const float* __restrict__ x0, _Float16* __restrict__ xh, int n) {
    int i = blockIdx.x * blockDim.x + threadIdx.x;     // over n*64
    if (i >= n * 64) return;
    int r = i >> 6, c = i & 63;
    xh[i] = (c < INF0) ? (_Float16)x0[r * INF0 + c] : (_Float16)0.f;
}
// W [K x 128] f32 -> Wt [128 x KP] f16 (transposed, zero-padded)
__global__ void cvt_wt(const float* __restrict__ W, _Float16* __restrict__ Wt, int K, int KP) {
    int i = blockIdx.x * blockDim.x + threadIdx.x;     // over 128*KP
    if (i >= 128 * KP) return;
    int col = i / KP, kk = i % KP;
    Wt[i] = (kk < K) ? (_Float16)W[kk * 128 + col] : (_Float16)0.f;
}

// ---------------- degree / norm ----------------
__global__ void deg_scatter(const int* __restrict__ dst, const float* __restrict__ ew,
                            float* __restrict__ deg, int nE) {
    int e = blockIdx.x * blockDim.x + threadIdx.x;
    if (e < nE) {
        float w = ew[e];
        if (w != 0.f) atomAddF(&deg[dst[e]], w);
    }
}
__global__ void rsqrt_inplace(float* p, int n) {
    int i = blockIdx.x * blockDim.x + threadIdx.x;
    if (i < n) p[i] = rsqrtf(p[i]);
}

// ---------------- WMMA GEMM: C[M x 128] = A[M x KP](f16) * Wt[128 x KP](f16,col-major) ----
template <int KP>
__global__ __launch_bounds__(256)
void gemm_wmma(const _Float16* __restrict__ A,
               const _Float16* __restrict__ Wt,
               const float* __restrict__ bias,   // nullable
               float* __restrict__ C) {
    const int m0   = blockIdx.x * 16;
    const int t    = threadIdx.x;
    const int lane = t & 31;
    const int wv   = t >> 5;                 // 8 waves -> 8 column tiles of 16
    const int colg = wv * 16 + (lane & 15);
    const int rowg = m0 + (lane & 15);
    const int hi   = lane >> 4;
    v8f acc = {};
    const _Float16* arow = A + (size_t)rowg * KP;
    const _Float16* bcol = Wt + (size_t)colg * KP;
#pragma unroll
    for (int k0 = 0; k0 < KP; k0 += 32) {
        v8h a0 = *(const v8h*)(arow + k0 + 8 * hi);
        v8h a1 = *(const v8h*)(arow + k0 + 16 + 8 * hi);
        v8h b0 = *(const v8h*)(bcol + k0 + 8 * hi);
        v8h b1 = *(const v8h*)(bcol + k0 + 16 + 8 * hi);
        v16h av = __builtin_shufflevector(a0, a1, 0, 1, 2, 3, 4, 5, 6, 7,
                                          8, 9, 10, 11, 12, 13, 14, 15);
        v16h bv = __builtin_shufflevector(b0, b1, 0, 1, 2, 3, 4, 5, 6, 7,
                                          8, 9, 10, 11, 12, 13, 14, 15);
        acc = __builtin_amdgcn_wmma_f32_16x16x32_f16(false, av, false, bv,
                                                     (short)0, acc, false, false);
    }
    // hoisted (uniform) bias read: one load instead of 8 branchy ones
    const float bb = (bias != nullptr) ? bias[colg] : 0.0f;
#pragma unroll
    for (int j = 0; j < 8; ++j) {
        int r = m0 + j + 8 * hi;
        C[(size_t)r * 128 + colg] = acc[j] + bb;
    }
}

// ---------------- edge scatters (one wave per edge, float4 per lane) ----------------
__global__ __launch_bounds__(256)
void conv_scatter(const int* __restrict__ src, const int* __restrict__ dst,
                  const float* __restrict__ ew, const float* __restrict__ dinv,
                  const float* __restrict__ xw, float* __restrict__ agg, int nE) {
    int g = blockIdx.x * blockDim.x + threadIdx.x;
    int e = g >> 5, lane = g & 31;
    if (e >= nE) return;
    float w = ew[e];
    if (w == 0.f) return;
    int s = src[e], d = dst[e];
    float nrm = dinv[s] * w * dinv[d];
    float4 v = ((const float4*)(xw + (size_t)s * 128))[lane];
    float* a = agg + (size_t)d * 128 + lane * 4;
    atomAddF(a + 0, v.x * nrm);
    atomAddF(a + 1, v.y * nrm);
    atomAddF(a + 2, v.z * nrm);
    atomAddF(a + 3, v.w * nrm);
}
__global__ __launch_bounds__(256)
void pool_scatter(const int* __restrict__ src, const int* __restrict__ dst,
                  const float* __restrict__ ew, const float* __restrict__ x,
                  float* __restrict__ aggP, int nE) {
    int g = blockIdx.x * blockDim.x + threadIdx.x;
    int e = g >> 5, lane = g & 31;
    if (e >= nE) return;
    float w = ew[e];
    if (w == 0.f) return;
    int s = src[e], d = dst[e];
    float4 v = ((const float4*)(x + (size_t)s * 128))[lane];
    float* a = aggP + (size_t)d * 128 + lane * 4;
    atomAddF(a + 0, v.x * w);
    atomAddF(a + 1, v.y * w);
    atomAddF(a + 2, v.z * w);
    atomAddF(a + 3, v.w * w);
}

// out(in place in agg) = relu(agg + xw * dinv^2 + bc)
__global__ __launch_bounds__(256)
void combine_relu(float* __restrict__ agg, const float* __restrict__ xw,
                  const float* __restrict__ dinv, const float* __restrict__ bc, int n) {
    int g = blockIdx.x * blockDim.x + threadIdx.x;
    int i = g >> 5, lane = g & 31;
    if (i >= n) return;
    float di = dinv[i]; float d2 = di * di;
    float4 a = ((const float4*)(agg + (size_t)i * 128))[lane];
    float4 xv = ((const float4*)(xw + (size_t)i * 128))[lane];
    float4 b = ((const float4*)bc)[lane];
    float4 o;
    o.x = fmaxf(a.x + xv.x * d2 + b.x, 0.f);
    o.y = fmaxf(a.y + xv.y * d2 + b.y, 0.f);
    o.z = fmaxf(a.z + xv.z * d2 + b.z, 0.f);
    o.w = fmaxf(a.w + xv.w * d2 + b.w, 0.f);
    ((float4*)(agg + (size_t)i * 128))[lane] = o;
}

// score[i] = aggP[i,:].Wrel + brel + x[i,:].Wroot  (wave per node)
__global__ __launch_bounds__(256)
void score_kernel(const float* __restrict__ aggP, const float* __restrict__ x,
                  const float* __restrict__ Wrel, const float* __restrict__ Wroot,
                  const float* __restrict__ brel, float* __restrict__ score, int n) {
    int g = blockIdx.x * blockDim.x + threadIdx.x;
    int i = g >> 5, lane = g & 31;
    if (i >= n) return;
    float4 a = ((const float4*)(aggP + (size_t)i * 128))[lane];
    float4 b = ((const float4*)(x + (size_t)i * 128))[lane];
    float4 wr = ((const float4*)Wrel)[lane];
    float4 wt = ((const float4*)Wroot)[lane];
    float p = a.x * wr.x + a.y * wr.y + a.z * wr.z + a.w * wr.w +
              b.x * wt.x + b.y * wt.y + b.z * wt.z + b.w * wt.w;
    for (int o = 16; o; o >>= 1) p += __shfl_xor(p, o, 32);
    if (lane == 0) score[i] = p + brel[0];
}

// ---------------- radix sort (descending score, stable => JAX top_k order) ----------------
__global__ void make_keys(const float* __restrict__ score, unsigned* __restrict__ keys,
                          unsigned* __restrict__ idx, int n) {
    int i = blockIdx.x * blockDim.x + threadIdx.x;
    if (i < n) {
        unsigned u = __float_as_uint(score[i]);
        unsigned asc = u ^ ((u >> 31) ? 0xFFFFFFFFu : 0x80000000u);
        keys[i] = ~asc;    // ascending key order == descending score
        idx[i] = (unsigned)i;
    }
}
__global__ __launch_bounds__(256)
void radix_hist(const unsigned* __restrict__ keys, unsigned* __restrict__ hist, int shift) {
    __shared__ unsigned sh[256];
    int t = threadIdx.x;
    sh[t] = 0;
    __syncthreads();
    int base = blockIdx.x * 4096;
    for (int r = 0; r < 16; ++r) {
        unsigned d = (keys[base + r * 256 + t] >> shift) & 255u;
        atomicAdd(&sh[d], 1u);
    }
    __syncthreads();
    hist[blockIdx.x * 256 + t] = sh[t];
}
__global__ __launch_bounds__(256)
void radix_scan(unsigned* __restrict__ hist, int numBlocks) {
    int d = threadIdx.x;
    unsigned run = 0;
    for (int b = 0; b < numBlocks; ++b) {
        unsigned c = hist[b * 256 + d];
        hist[b * 256 + d] = run;
        run += c;
    }
    __shared__ unsigned sA[256], sB[256];
    sA[d] = run;
    __syncthreads();
    unsigned *s = sA, *t = sB;
    for (int off = 1; off < 256; off <<= 1) {
        t[d] = s[d] + ((d >= off) ? s[d - off] : 0u);
        __syncthreads();
        unsigned* tmp = s; s = t; t = tmp;
    }
    unsigned excl = (d == 0) ? 0u : s[d - 1];
    __syncthreads();
    for (int b = 0; b < numBlocks; ++b) hist[b * 256 + d] += excl;
}
__global__ __launch_bounds__(256)
void radix_scatter(const unsigned* __restrict__ keysIn, const unsigned* __restrict__ idxIn,
                   unsigned* __restrict__ keysOut, unsigned* __restrict__ idxOut,
                   const unsigned* __restrict__ blockOff, int shift) {
    __shared__ unsigned cursor[256];
    __shared__ unsigned wcnt[8][256];
    int t = threadIdx.x, lane = t & 31, wv = t >> 5;
    cursor[t] = blockOff[blockIdx.x * 256 + t];
    int base = blockIdx.x * 4096;
    for (int r = 0; r < 16; ++r) {
#pragma unroll
        for (int ww = 0; ww < 8; ++ww) wcnt[ww][t] = 0;
        __syncthreads();
        int item = base + r * 256 + t;
        unsigned key = keysIn[item];
        unsigned dig = (key >> shift) & 255u;
        unsigned same = 0xFFFFFFFFu;
#pragma unroll
        for (int b = 0; b < 8; ++b) {
            unsigned v = (unsigned)__ballot((dig >> b) & 1);
            same &= ((dig >> b) & 1) ? v : ~v;
        }
        unsigned below = (lane == 0) ? 0u : (0xFFFFFFFFu >> (32 - lane));
        int wrank = __popc(same & below);
        if (wrank == 0) wcnt[wv][dig] = (unsigned)__popc(same);
        __syncthreads();
        int rank = wrank;
        for (int ww = 0; ww < wv; ++ww) rank += (int)wcnt[ww][dig];
        unsigned pos = cursor[dig] + (unsigned)rank;
        keysOut[pos] = key;
        idxOut[pos] = idxIn[item];
        __syncthreads();
        unsigned tot = 0;
#pragma unroll
        for (int ww = 0; ww < 8; ++ww) tot += wcnt[ww][t];
        cursor[t] += tot;
        __syncthreads();
    }
}

// ---------------- pooling epilogue ----------------
__global__ void set_mapping(const unsigned* __restrict__ idx, int* __restrict__ mapping, int k) {
    int j = blockIdx.x * blockDim.x + threadIdx.x;
    if (j < k) mapping[idx[j]] = j;
}
// xkh (f16) = x[perm] * tanh(score[perm]) ; feeds next layer's WMMA GEMM directly
__global__ __launch_bounds__(256)
void gather_scale(const float* __restrict__ x, const unsigned* __restrict__ idx,
                  const float* __restrict__ score, _Float16* __restrict__ xkh, int k) {
    int g = blockIdx.x * blockDim.x + threadIdx.x;
    int j = g >> 5, lane = g & 31;
    if (j >= k) return;
    unsigned node = idx[j];
    float gsc = tanhf(score[node]);
    float4 v = ((const float4*)(x + (size_t)node * 128))[lane];
    v4h h;
    h[0] = (_Float16)(v.x * gsc);
    h[1] = (_Float16)(v.y * gsc);
    h[2] = (_Float16)(v.z * gsc);
    h[3] = (_Float16)(v.w * gsc);
    ((v4h*)(xkh + (size_t)j * 128))[lane] = h;
}
__global__ void relabel(const int* __restrict__ src, const int* __restrict__ dst,
                        const float* __restrict__ ew, const int* __restrict__ mapping,
                        int* __restrict__ nsrc, int* __restrict__ ndst,
                        float* __restrict__ newew, int nE) {
    int e = blockIdx.x * blockDim.x + threadIdx.x;
    if (e >= nE) return;
    int s = src[e], d = dst[e];
    float w = ew[e];
    int ns = mapping[s], nd = mapping[d];
    bool valid = (ns >= 0) && (nd >= 0) && (w > 0.f);
    newew[e] = valid ? w : 0.f;
    nsrc[e] = valid ? ns : 0;
    ndst[e] = valid ? nd : 0;
}

// ---------------- log_softmax over 128 cols (wave per row, in place) ----------------
__global__ __launch_bounds__(256)
void log_softmax_rows(float* __restrict__ out, int n) {
    int g = blockIdx.x * blockDim.x + threadIdx.x;
    int i = g >> 5, lane = g & 31;
    if (i >= n) return;
    float4 v = ((const float4*)(out + (size_t)i * 128))[lane];
    float m = fmaxf(fmaxf(v.x, v.y), fmaxf(v.z, v.w));
    for (int o = 16; o; o >>= 1) m = fmaxf(m, __shfl_xor(m, o, 32));
    float s = expf(v.x - m) + expf(v.y - m) + expf(v.z - m) + expf(v.w - m);
    for (int o = 16; o; o >>= 1) s += __shfl_xor(s, o, 32);
    float l = m + logf(s);
    v.x -= l; v.y -= l; v.z -= l; v.w -= l;
    ((float4*)(out + (size_t)i * 128))[lane] = v;
}

// ---------------- host orchestration ----------------
extern "C" void kernel_launch(void* const* d_in, const int* in_sizes, int n_in,
                              void* d_out, int out_size, void* d_ws, size_t ws_size,
                              hipStream_t stream) {
    (void)in_sizes; (void)n_in; (void)ws_size; (void)out_size;
    const float* x0 = (const float*)d_in[0];
    const int*   ei = (const int*)d_in[1];
    const float *Wc[NLAY], *bc[NLAY], *Wrel[NLAY], *brel[NLAY], *Wroot[NLAY];
    int p = 2;
    for (int l = 0; l < NLAY; ++l) {
        Wc[l]    = (const float*)d_in[p++];
        bc[l]    = (const float*)d_in[p++];
        Wrel[l]  = (const float*)d_in[p++];
        brel[l]  = (const float*)d_in[p++];
        Wroot[l] = (const float*)d_in[p++];
    }
    const float* linW = (const float*)d_in[p++];
    const float* linb = (const float*)d_in[p++];

    // workspace carve-out
    char* wbase = (char*)d_ws;
    size_t off = 0;
    auto take = [&](size_t bytes) -> char* {
        char* q = wbase + off;
        off += (bytes + 255) & ~(size_t)255;
        return q;
    };
    float*     xw     = (float*)take((size_t)N0 * 128 * 4);       // also pool accumulator
    float*     agg    = (float*)take((size_t)N0 * 128 * 4);       // conv output in place
    _Float16*  xh0    = (_Float16*)take((size_t)N0 * 64 * 2);     // padded f16 input
    _Float16*  poolAh = (_Float16*)take((size_t)(N0 / 2) * 128 * 2);
    _Float16*  poolBh = (_Float16*)take((size_t)(N0 / 2) * 128 * 2);
    _Float16*  Wth    = (_Float16*)take((size_t)128 * 128 * 2);   // transposed f16 weights
    float*     deg    = (float*)take((size_t)N0 * 4);             // becomes dinv
    float*     score  = (float*)take((size_t)N0 * 4);
    unsigned*  keysA  = (unsigned*)take((size_t)N0 * 4);
    unsigned*  idxA   = (unsigned*)take((size_t)N0 * 4);
    unsigned*  keysB  = (unsigned*)take((size_t)N0 * 4);
    unsigned*  idxB   = (unsigned*)take((size_t)N0 * 4);
    int*       mapng  = (int*)take((size_t)N0 * 4);
    unsigned*  hist   = (unsigned*)take(32 * 256 * 4);
    int*       srcA   = (int*)take((size_t)NE * 4);
    int*       dstA   = (int*)take((size_t)NE * 4);
    float*     ewA    = (float*)take((size_t)NE * 4);
    int*       srcB   = (int*)take((size_t)NE * 4);
    int*       dstB   = (int*)take((size_t)NE * 4);
    float*     ewB    = (float*)take((size_t)NE * 4);

    const dim3 B256(256);
    // layer-0 edge weights are all ones; pad+convert x0 once
    fill_f32<<<2048, B256, 0, stream>>>(ewA, 1.0f, NE);
    cvt_x0<<<(N0 * 64) / 256, B256, 0, stream>>>(x0, xh0, N0);

    int n = N0;
    const _Float16* xin = xh0;    // f16 GEMM input, KP-padded
    int kp = 64;
    const int* src = ei;
    const int* dst = ei + NE;
    const float* ew = ewA;

    for (int l = 0; l < NLAY; ++l) {
        const int ewaves = (NE + 7) / 8;     // wave-per-edge grid (256 thr = 8 waves)
        const int nwaves = (n + 7) / 8;
        // 1) symmetric normalization
        fill_f32<<<2048, B256, 0, stream>>>(deg, 1.0f, n);
        deg_scatter<<<NE / 256, B256, 0, stream>>>(dst, ew, deg, NE);
        rsqrt_inplace<<<(n + 255) / 256, B256, 0, stream>>>(deg, n);
        // 2) xw = xin @ Wc   (WMMA, f16 operands, fully unrolled K)
        cvt_wt<<<(128 * kp) / 256, B256, 0, stream>>>(Wc[l], Wth, (l == 0) ? INF0 : HDIM, kp);
        if (kp == 64)
            gemm_wmma<64><<<n / 16, B256, 0, stream>>>(xin, Wth, nullptr, xw);
        else
            gemm_wmma<128><<<n / 16, B256, 0, stream>>>(xin, Wth, nullptr, xw);
        // 3) neighbor aggregation
        fill_f32<<<4096, B256, 0, stream>>>(agg, 0.f, n * 128);
        conv_scatter<<<ewaves, B256, 0, stream>>>(src, dst, ew, deg, xw, agg, NE);
        combine_relu<<<nwaves, B256, 0, stream>>>(agg, xw, deg, bc[l], n);
        // 4) SAG pool: GraphConv score
        fill_f32<<<4096, B256, 0, stream>>>(xw, 0.f, n * 128);   // xw reused as aggP
        pool_scatter<<<ewaves, B256, 0, stream>>>(src, dst, ew, agg, xw, NE);
        score_kernel<<<nwaves, B256, 0, stream>>>(xw, agg, Wrel[l], Wroot[l], brel[l], score, n);
        // 5) stable descending sort of scores (4x 8-bit LSD radix)
        make_keys<<<(n + 255) / 256, B256, 0, stream>>>(score, keysA, idxA, n);
        int nb = n / 4096;
        unsigned *kin = keysA, *iin = idxA, *kout = keysB, *iout = idxB;
        for (int pass = 0; pass < 4; ++pass) {
            int shift = pass * 8;
            radix_hist<<<nb, B256, 0, stream>>>(kin, hist, shift);
            radix_scan<<<1, B256, 0, stream>>>(hist, nb);
            radix_scatter<<<nb, B256, 0, stream>>>(kin, iin, kout, iout, hist, shift);
            unsigned* tk = kin; kin = kout; kout = tk;
            unsigned* ti = iin; iin = iout; iout = ti;
        }
        const unsigned* sidx = iin;          // sorted indices after 4 passes
        int k = n / 2;
        fill_i32<<<(n + 255) / 256, B256, 0, stream>>>(mapng, -1, n);
        set_mapping<<<(k + 255) / 256, B256, 0, stream>>>(sidx, mapng, k);
        _Float16* xk = (l & 1) ? poolBh : poolAh;
        gather_scale<<<(k + 7) / 8, B256, 0, stream>>>(agg, sidx, score, xk, k);
        // 6) relabel edges into the other edge set
        int* nsrc = (l & 1) ? srcA : srcB;
        int* ndst = (l & 1) ? dstA : dstB;
        float* nw = (l & 1) ? ewA : ewB;
        relabel<<<NE / 256, B256, 0, stream>>>(src, dst, ew, mapng, nsrc, ndst, nw, NE);
        // advance
        xin = xk; kp = HDIM; n = k;
        src = nsrc; dst = ndst; ew = nw;
    }

    // final linear + log_softmax  (n == 8192, out_size == 8192*128)
    float* out = (float*)d_out;
    cvt_wt<<<(128 * 128) / 256, B256, 0, stream>>>(linW, Wth, HDIM, HDIM);
    gemm_wmma<128><<<n / 16, B256, 0, stream>>>(xin, Wth, linb, out);
    log_softmax_rows<<<(n + 7) / 8, B256, 0, stream>>>(out, n);
}
